// MatcherPair_9251359556104
// MI455X (gfx1250) — matrix operand order repaired and checked by reading, not compiled, not run on hardware
//
#include <hip/hip_runtime.h>
#include <hip/hip_bf16.h>
#include <math.h>

typedef __attribute__((ext_vector_type(16))) _Float16 v16h;
typedef __attribute__((ext_vector_type(8)))  _Float16 v8h;
typedef __attribute__((ext_vector_type(8)))  float    v8f;

// ---------------------------------------------------------------------------
// WMMA fragment helpers (wave32; A/B in contiguous k order chosen here, C/D per CDNA5 ISA 7.12.2)
// A (16x32 f16): lanes 0-15 row M=lane, halves K0..15; lanes 16-31 K16..31
// B (32x16 f16): lanes 0-15 col N=lane, halves K0..15; lanes 16-31 K16..31
// C/D (16x16 f32): lane l -> N=l&15, VGPR v -> M = v + (l<16 ? 0 : 8)
// ---------------------------------------------------------------------------
__device__ inline v16h frag_a(const _Float16* p, int lane) {
  return *(const v16h*)(p + ((lane & 16) ? 16 : 0));
}
__device__ inline v16h frag_b(const _Float16* p, int lane) {
  return *(const v16h*)(p + ((lane & 16) ? 16 : 0));
}
__device__ inline v8f wmma16(v16h a, v16h b, v8f c) {
  return __builtin_amdgcn_wmma_f32_16x16x32_f16(false, a, false, b, (short)0, c,
                                                false, false);
}
__device__ inline v8f zero8f() {
  v8f z = {0.f, 0.f, 0.f, 0.f, 0.f, 0.f, 0.f, 0.f};
  return z;
}

// sortable-unsigned mapping for float atomicMax
__device__ inline unsigned ordf(float f) {
  int i = __float_as_int(f);
  unsigned u = (unsigned)i;
  return (i < 0) ? ~u : (u | 0x80000000u);
}
__device__ inline float unordf(unsigned u) {
  int i = (u & 0x80000000u) ? (int)(u & 0x7fffffffu) : (int)~u;
  return __int_as_float(i);
}

__device__ inline void ins5(float* t, float v) {
  if (v <= t[4]) return;
  t[4] = v;
#pragma unroll
  for (int k = 4; k > 0; --k) {
    if (t[k] > t[k - 1]) { float tmp = t[k]; t[k] = t[k - 1]; t[k - 1] = tmp; }
  }
}

__device__ inline float cubw(float x) {
  const float A = -0.75f;
  x = fabsf(x);
  if (x <= 1.f) return ((A + 2.f) * x - (A + 3.f)) * x * x + 1.f;
  if (x < 2.f)  return ((A * x - 5.f * A) * x + 8.f * A) * x - 4.f * A;
  return 0.f;
}

// ---------------------------------------------------------------------------
// Small utility kernels
// ---------------------------------------------------------------------------
__global__ void fill_zero(unsigned* __restrict__ p, int n) {
  int t = blockIdx.x * blockDim.x + threadIdx.x;
  if (t < n) p[t] = 0u;
}

// cubic interpolation tables (align_corners=True, clamped taps)
__global__ void cubic_table(int out_s, int in_s, int* __restrict__ idx,
                            float* __restrict__ wt) {
  int i = blockIdx.x * blockDim.x + threadIdx.x;
  if (i >= out_s) return;
  float s = (out_s > 1) ? (float)i * (float)(in_s - 1) / (float)(out_s - 1) : 0.f;
  float f = floorf(s);
  float t = s - f;
  int fi = (int)f;
#pragma unroll
  for (int o = -1; o <= 2; ++o) {
    int id = fi + o;
    id = id < 0 ? 0 : (id > in_s - 1 ? in_s - 1 : id);
    idx[i * 4 + (o + 1)] = id;
    wt[i * 4 + (o + 1)] = cubw(t - (float)o);
  }
}

// OIHW f32 -> [tap][co][ciPad] f16 (zero padded channels)
__global__ void pack_w(const float* __restrict__ w, _Float16* __restrict__ wp,
                       int Co, int Ci, int CiPad) {
  int t = blockIdx.x * blockDim.x + threadIdx.x;
  int total = 9 * Co * CiPad;
  if (t >= total) return;
  int ci  = t % CiPad;
  int co  = (t / CiPad) % Co;
  int tap = t / (CiPad * Co);
  float v = (ci < Ci) ? w[(co * Ci + ci) * 9 + tap] : 0.f;
  wp[t] = (_Float16)v;
}

// NCHW f32 (3,H,W) -> zero-haloed NHWC f16 (H+2)x(W+2)x32
__global__ void pack_input(const float* __restrict__ x, _Float16* __restrict__ o,
                           int H, int W) {
  int t = blockIdx.x * blockDim.x + threadIdx.x;
  int total = (H + 2) * (W + 2) * 32;
  if (t >= total) return;
  int c  = t % 32;
  int px = (t / 32) % (W + 2);
  int py = t / (32 * (W + 2));
  float v = 0.f;
  if (c < 3 && py >= 1 && py <= H && px >= 1 && px <= W)
    v = x[c * H * W + (py - 1) * W + (px - 1)];
  o[t] = (_Float16)v;
}

// ---------------------------------------------------------------------------
// 3x3 conv as 9 tap-GEMMs. One wave computes 16 cout x (16*NS) pixels.
// inPad/outPad: zero-haloed NHWC f16 (H+2)x(W+2)xC. wpack: [9][Cout][Cin] f16.
// ---------------------------------------------------------------------------
template <int NS>
__global__ void __launch_bounds__(32)
conv3x3_wmma(const _Float16* __restrict__ inPad, const _Float16* __restrict__ wpack,
             const float* __restrict__ bias, _Float16* __restrict__ outPad,
             int H, int W, int Cin, int Cout, int relu) {
  const int lane = threadIdx.x;
  const int TW = 16 * NS;
  const int xTiles = W / TW;
  const int xt = blockIdx.x % xTiles;
  const int y  = blockIdx.x / xTiles;
  const int co0 = blockIdx.y * 16;
  const int x0 = xt * TW;
  const int nLane = lane & 15;
  v8f acc[NS];
#pragma unroll
  for (int s = 0; s < NS; ++s) acc[s] = zero8f();
  const int aRow = (co0 + nLane) * Cin;
  for (int t = 0; t < 9; ++t) {
    const int ty = t / 3, tx = t % 3;
    const _Float16* wp = wpack + (size_t)t * Cout * Cin + aRow;
    const _Float16* ip = inPad + ((size_t)(y + ty) * (W + 2) + (x0 + tx)) * Cin;
    for (int k0 = 0; k0 < Cin; k0 += 32) {
      v16h a = frag_a(wp + k0, lane);
#pragma unroll
      for (int s = 0; s < NS; ++s) {
        v16h b = frag_b(ip + (size_t)(s * 16 + nLane) * Cin + k0, lane);
        acc[s] = wmma16(a, b, acc[s]);
      }
    }
  }
  const int coBase = co0 + ((lane & 16) ? 8 : 0);
  float bv[8];
#pragma unroll
  for (int v = 0; v < 8; ++v) bv[v] = bias[coBase + v];
#pragma unroll
  for (int s = 0; s < NS; ++s) {
    const int xo = x0 + s * 16 + nLane;
    _Float16* op = outPad + ((size_t)(y + 1) * (W + 2) + (xo + 1)) * Cout + coBase;
    v8h h;
#pragma unroll
    for (int v = 0; v < 8; ++v) {
      float f = acc[s][v] + bv[v];
      if (relu) f = fmaxf(f, 0.f);
      h[v] = (_Float16)f;
    }
    *(v8h*)op = h;
  }
}

// 2x2 max pool on haloed NHWC f16 buffers (writes interior of out)
__global__ void pool2(const _Float16* __restrict__ in, _Float16* __restrict__ out,
                      int Ho, int Wo, int C) {
  int t = blockIdx.x * blockDim.x + threadIdx.x;
  int total = Ho * Wo * C;
  if (t >= total) return;
  int c = t % C;
  int x = (t / C) % Wo;
  int y = t / (C * Wo);
  int Wi = 2 * Wo;
  const _Float16* b = in + ((size_t)(2 * y + 1) * (Wi + 2) + (2 * x + 1)) * C + c;
  float a0 = (float)b[0];
  float a1 = (float)b[C];
  float a2 = (float)b[(size_t)(Wi + 2) * C];
  float a3 = (float)b[(size_t)(Wi + 2) * C + C];
  float m = fmaxf(fmaxf(a0, a1), fmaxf(a2, a3));
  out[((size_t)(y + 1) * (Wo + 2) + (x + 1)) * C + c] = (_Float16)m;
}

// bicubic upsample of haloed NHWC f16 (interior Wi x Wi, C ch) -> Ho x Wo x C f16
__global__ void bicubic_up_feat(const _Float16* __restrict__ inPad,
                                _Float16* __restrict__ out,
                                const int* __restrict__ idxT,
                                const float* __restrict__ wT,
                                int Wi, int C, int Ho, int Wo) {
  int t = blockIdx.x * blockDim.x + threadIdx.x;
  int cv = C / 8;
  int total = Ho * Wo * cv;
  if (t >= total) return;
  int c8 = t % cv;
  int X = (t / cv) % Wo;
  int Y = t / (cv * Wo);
  float acc[8];
#pragma unroll
  for (int k = 0; k < 8; ++k) acc[k] = 0.f;
#pragma unroll
  for (int a = 0; a < 4; ++a) {
    int iy = idxT[Y * 4 + a];
    float wy = wT[Y * 4 + a];
#pragma unroll
    for (int b = 0; b < 4; ++b) {
      int ix = idxT[X * 4 + b];
      float wgt = wy * wT[X * 4 + b];
      const v8h v = *(const v8h*)(inPad +
          ((size_t)(iy + 1) * (Wi + 2) + (ix + 1)) * C + c8 * 8);
#pragma unroll
      for (int k = 0; k < 8; ++k) acc[k] += wgt * (float)v[k];
    }
  }
  v8h o;
#pragma unroll
  for (int k = 0; k < 8; ++k) o[k] = (_Float16)acc[k];
  *(v8h*)(out + ((size_t)(Y * Wo + X) * C) + c8 * 8) = o;
}

// concat [64ch feat1 interior | 256ch up] + L2-normalize -> [pixel][320] f16
__global__ void concat_norm(const _Float16* __restrict__ f1Pad,
                            const _Float16* __restrict__ up,
                            _Float16* __restrict__ outT, int H, int W) {
  int p = blockIdx.x * blockDim.x + threadIdx.x;
  if (p >= H * W) return;
  int y = p / W, x = p % W;
  const _Float16* a = f1Pad + ((size_t)(y + 1) * (W + 2) + (x + 1)) * 64;
  const _Float16* b = up + (size_t)p * 256;
  float ss = 0.f;
  for (int i = 0; i < 64; ++i) { float v = (float)a[i]; ss += v * v; }
  for (int i = 0; i < 256; ++i) { float v = (float)b[i]; ss += v * v; }
  float sc = 1.f / (1e-8f + sqrtf(ss));
  _Float16* o = outT + (size_t)p * 320;
  for (int i = 0; i < 64; ++i)  o[i] = (_Float16)((float)a[i] * sc);
  for (int i = 0; i < 256; ++i) o[64 + i] = (_Float16)((float)b[i] * sc);
}

// exact 2x2 mean: [16384][320] -> [4096][320] f16
__global__ void down2_mean(const _Float16* __restrict__ in,
                           _Float16* __restrict__ out) {
  int t = blockIdx.x * blockDim.x + threadIdx.x;
  if (t >= 4096 * 320) return;
  int c = t % 320;
  int mn = t / 320;
  int m = mn / 64, n = mn % 64;
  size_t p00 = ((size_t)(2 * m) * 128 + 2 * n) * 320 + c;
  float s = (float)in[p00] + (float)in[p00 + 320] +
            (float)in[p00 + 128 * 320] + (float)in[p00 + 128 * 320 + 320];
  out[t] = (_Float16)(0.25f * s);
}

// ---------------------------------------------------------------------------
// Correlation passes. dist[i][j] = <frefT[i], ftemT[j]> / 320
// phase 0: column maxes (over i) of coef_ref*d  -> M1u[j]
//          row    maxes (over j) of coef_temp*d -> M2u[i]
// phase 1: exp-sums S1[j], S2[i]
// ---------------------------------------------------------------------------
__global__ void __launch_bounds__(32)
corr_stats(const _Float16* __restrict__ frefT, const _Float16* __restrict__ ftemT,
           const float* __restrict__ coefR, const float* __restrict__ coefT,
           unsigned* __restrict__ M1u, unsigned* __restrict__ M2u,
           float* __restrict__ S1, float* __restrict__ S2, int phase) {
  __shared__ float tile[16 * 64];
  const int lane = threadIdx.x;
  const int i0 = blockIdx.y * 16, j0 = blockIdx.x * 64;
  const int nLane = lane & 15;
  v8f acc[4];
#pragma unroll
  for (int s = 0; s < 4; ++s) acc[s] = zero8f();
  const _Float16* ar = frefT + (size_t)(i0 + nLane) * 320;
#pragma unroll
  for (int k0 = 0; k0 < 320; k0 += 32) {
    v16h a = frag_a(ar + k0, lane);
#pragma unroll
    for (int s = 0; s < 4; ++s) {
      v16h b = frag_b(ftemT + (size_t)(j0 + s * 16 + nLane) * 320 + k0, lane);
      acc[s] = wmma16(a, b, acc[s]);
    }
  }
  const int mBase = (lane & 16) ? 8 : 0;
#pragma unroll
  for (int s = 0; s < 4; ++s)
#pragma unroll
    for (int v = 0; v < 8; ++v)
      tile[(mBase + v) * 64 + s * 16 + nLane] = acc[s][v] * (1.0f / 320.0f);
  __syncthreads();
  const float cr = *coefR, ct = *coefT;
  if (phase == 0) {
    if (lane < 16) {
      float m = -1e30f;
      for (int c = 0; c < 64; ++c) m = fmaxf(m, ct * tile[lane * 64 + c]);
      atomicMax(&M2u[i0 + lane], ordf(m));
    }
    for (int c = lane; c < 64; c += 32) {
      float m = -1e30f;
      for (int r = 0; r < 16; ++r) m = fmaxf(m, cr * tile[r * 64 + c]);
      atomicMax(&M1u[j0 + c], ordf(m));
    }
  } else {
    if (lane < 16) {
      float M = unordf(M2u[i0 + lane]);
      float s = 0.f;
      for (int c = 0; c < 64; ++c) s += expf(ct * tile[lane * 64 + c] - M);
      atomicAdd(&S2[i0 + lane], s);
    }
    for (int c = lane; c < 64; c += 32) {
      float M = unordf(M1u[j0 + c]);
      float s = 0.f;
      for (int r = 0; r < 16; ++r) s += expf(cr * tile[r * 64 + c] - M);
      atomicAdd(&S1[j0 + c], s);
    }
  }
}

__global__ void finalize_terms(const unsigned* __restrict__ M1u,
                               const float* __restrict__ S1,
                               const unsigned* __restrict__ M2u,
                               const float* __restrict__ S2,
                               float* __restrict__ c1, float* __restrict__ r2) {
  int t = blockIdx.x * blockDim.x + threadIdx.x;
  if (t < 16384) c1[t] = 0.5f * unordf(M1u[t]) + 0.5f * logf(S1[t]);
  if (t < 4096)  r2[t] = 0.5f * unordf(M2u[t]) + 0.5f * logf(S2[t]);
}

// top-5 per row i of confidence = exp(key - r2[i]); key = hs*d - c1[j]
// block: 16 rows, 4 waves each scanning a 4096-column stripe.
// A fragments (10 K-steps) are register-resident across all 256 j-tiles;
// next tile's B row is prefetched (global_prefetch) to hide load latency.
__global__ void __launch_bounds__(128)
corr_topk(const _Float16* __restrict__ frefT, const _Float16* __restrict__ ftemT,
          const float* __restrict__ coefR, const float* __restrict__ coefT,
          const float* __restrict__ c1, const float* __restrict__ r2,
          float* __restrict__ conf) {
  __shared__ float tile[4][16 * 16];
  __shared__ float mg[4][16][5];
  const int lane = threadIdx.x & 31;
  const int w = threadIdx.x >> 5;
  const int i0 = blockIdx.x * 16;
  const float hs = 0.5f * ((*coefR) + (*coefT)) * (1.0f / 320.0f);
  const int nLane = lane & 15;
  const int mBase = (lane & 16) ? 8 : 0;
  float top[5];
#pragma unroll
  for (int q = 0; q < 5; ++q) top[q] = -1e30f;

  // register-cache the 16x320 A stripe (10 fragments, reused for 256 tiles)
  const _Float16* ar = frefT + (size_t)(i0 + nLane) * 320;
  v16h af[10];
#pragma unroll
  for (int kk = 0; kk < 10; ++kk) af[kk] = frag_a(ar + kk * 32, lane);

  for (int jt = 0; jt < 256; ++jt) {
    const int j0 = w * 4096 + jt * 16;
    const _Float16* br = ftemT + (size_t)(j0 + nLane) * 320;
    if (jt < 255) __builtin_prefetch(br + 16 * 320, 0, 3);  // next tile's B row
    v8f acc = zero8f();
#pragma unroll
    for (int kk = 0; kk < 10; ++kk) {
      v16h b = frag_b(br + kk * 32, lane);
      acc = wmma16(af[kk], b, acc);
    }
    const float cj = c1[j0 + nLane];
#pragma unroll
    for (int v = 0; v < 8; ++v)
      tile[w][(mBase + v) * 16 + nLane] = hs * acc[v] - cj;
    __syncthreads();
    if (lane < 16) {
#pragma unroll
      for (int c = 0; c < 16; ++c) ins5(top, tile[w][lane * 16 + c]);
    }
    __syncthreads();
  }
  if (lane < 16) {
#pragma unroll
    for (int q = 0; q < 5; ++q) mg[w][lane][q] = top[q];
  }
  __syncthreads();
  if (w == 0 && lane < 16) {
    float t5[5];
#pragma unroll
    for (int q = 0; q < 5; ++q) t5[q] = -1e30f;
    for (int ww = 0; ww < 4; ++ww)
      for (int q = 0; q < 5; ++q) ins5(t5, mg[ww][lane][q]);
    const float rr = r2[i0 + lane];
#pragma unroll
    for (int q = 0; q < 5; ++q)
      conf[q * 4096 + i0 + lane] = expf(t5[q] - rr);
  }
}

// tiny NCHW f32 head convs (3x3 pad1 or 1x1 pad0)
__global__ void conv_head(const float* __restrict__ in, const float* __restrict__ w,
                          const float* __restrict__ b, float* __restrict__ out,
                          int Cin, int Cout, int H, int W, int K, int relu) {
  int t = blockIdx.x * blockDim.x + threadIdx.x;
  int total = Cout * H * W;
  if (t >= total) return;
  int x = t % W;
  int y = (t / W) % H;
  int co = t / (W * H);
  float s = b[co];
  if (K == 3) {
    for (int ci = 0; ci < Cin; ++ci)
      for (int ty = 0; ty < 3; ++ty) {
        int yy = y + ty - 1;
        if (yy < 0 || yy >= H) continue;
        for (int tx = 0; tx < 3; ++tx) {
          int xx = x + tx - 1;
          if (xx < 0 || xx >= W) continue;
          s += w[((co * Cin + ci) * 3 + ty) * 3 + tx] * in[(ci * H + yy) * W + xx];
        }
      }
  } else {
    for (int ci = 0; ci < Cin; ++ci) s += w[co * Cin + ci] * in[(ci * H + y) * W + x];
  }
  if (relu) s = fmaxf(s, 0.f);
  out[t] = s;
}

// single-channel bicubic upsample f32 -> f32
__global__ void bicubic_up1(const float* __restrict__ in, float* __restrict__ out,
                            const int* __restrict__ idxT, const float* __restrict__ wT,
                            int Wi, int Ho, int Wo) {
  int t = blockIdx.x * blockDim.x + threadIdx.x;
  if (t >= Ho * Wo) return;
  int X = t % Wo, Y = t / Wo;
  float acc = 0.f;
#pragma unroll
  for (int a = 0; a < 4; ++a) {
    int iy = idxT[Y * 4 + a];
    float wy = wT[Y * 4 + a];
#pragma unroll
    for (int bb = 0; bb < 4; ++bb) {
      int ix = idxT[X * 4 + bb];
      acc += wy * wT[X * 4 + bb] * in[iy * Wi + ix];
    }
  }
  out[t] = acc;
}

// ---------------------------------------------------------------------------
static inline int cdiv(int a, int b) { return (a + b - 1) / b; }

extern "C" void kernel_launch(void* const* d_in, const int* in_sizes, int n_in,
                              void* d_out, int out_size, void* d_ws, size_t ws_size,
                              hipStream_t stream) {
  const float* x_ref = (const float*)d_in[0];
  const float* x_tem = (const float*)d_in[1];
  const float* W0  = (const float*)d_in[2];  const float* B0  = (const float*)d_in[3];
  const float* W2  = (const float*)d_in[4];  const float* B2  = (const float*)d_in[5];
  const float* W5  = (const float*)d_in[6];  const float* B5  = (const float*)d_in[7];
  const float* W7  = (const float*)d_in[8];  const float* B7  = (const float*)d_in[9];
  const float* W10 = (const float*)d_in[10]; const float* B10 = (const float*)d_in[11];
  const float* W12 = (const float*)d_in[12]; const float* B12 = (const float*)d_in[13];
  const float* W14 = (const float*)d_in[14]; const float* B14 = (const float*)d_in[15];
  const float* coefR = (const float*)d_in[16];
  const float* coefT = (const float*)d_in[17];
  const float* FW1 = (const float*)d_in[18]; const float* FB1 = (const float*)d_in[19];
  const float* FW2 = (const float*)d_in[20]; const float* FB2 = (const float*)d_in[21];
  const float* FW3 = (const float*)d_in[22]; const float* FB3 = (const float*)d_in[23];

  char* p = (char*)d_ws;
  auto alloc = [&](size_t bytes) -> char* {
    char* r = p;
    p += (bytes + 255) & ~(size_t)255;
    return r;
  };

  _Float16* wp0  = (_Float16*)alloc((size_t)9 * 64 * 32 * 2);
  _Float16* wp2  = (_Float16*)alloc((size_t)9 * 64 * 64 * 2);
  _Float16* wp5  = (_Float16*)alloc((size_t)9 * 128 * 64 * 2);
  _Float16* wp7  = (_Float16*)alloc((size_t)9 * 128 * 128 * 2);
  _Float16* wp10 = (_Float16*)alloc((size_t)9 * 256 * 128 * 2);
  _Float16* wp12 = (_Float16*)alloc((size_t)9 * 256 * 256 * 2);
  _Float16* wp14 = (_Float16*)alloc((size_t)9 * 256 * 256 * 2);
  _Float16* inPad = (_Float16*)alloc((size_t)130 * 130 * 32 * 2);

  char* convZeroBase = p;
  _Float16* f1Pad = (_Float16*)alloc((size_t)130 * 130 * 64 * 2);
  _Float16* cA    = (_Float16*)alloc((size_t)130 * 130 * 64 * 2);
  _Float16* p1    = (_Float16*)alloc((size_t)66 * 66 * 64 * 2);
  _Float16* cB    = (_Float16*)alloc((size_t)66 * 66 * 128 * 2);
  _Float16* cC    = (_Float16*)alloc((size_t)66 * 66 * 128 * 2);
  _Float16* p2    = (_Float16*)alloc((size_t)34 * 34 * 128 * 2);
  _Float16* cD    = (_Float16*)alloc((size_t)34 * 34 * 256 * 2);
  _Float16* cE    = (_Float16*)alloc((size_t)34 * 34 * 256 * 2);
  _Float16* cF    = (_Float16*)alloc((size_t)34 * 34 * 256 * 2);
  size_t convZeroBytes = (size_t)(p - convZeroBase);

  _Float16* up    = (_Float16*)alloc((size_t)128 * 128 * 256 * 2);
  _Float16* frefT = (_Float16*)alloc((size_t)4096 * 320 * 2);
  _Float16* ftemT = (_Float16*)alloc((size_t)16384 * 320 * 2);

  char* statsBase = p;
  unsigned* M1u = (unsigned*)alloc((size_t)16384 * 4);
  float*    S1  = (float*)alloc((size_t)16384 * 4);
  unsigned* M2u = (unsigned*)alloc((size_t)4096 * 4);
  float*    S2  = (float*)alloc((size_t)4096 * 4);
  size_t statsBytes = (size_t)(p - statsBase);

  float* c1   = (float*)alloc((size_t)16384 * 4);
  float* r2   = (float*)alloc((size_t)4096 * 4);
  float* conf = (float*)alloc((size_t)5 * 4096 * 4);
  float* h1   = (float*)alloc((size_t)8 * 4096 * 4);
  float* h2   = (float*)alloc((size_t)8 * 4096 * 4);
  float* h3   = (float*)alloc((size_t)4096 * 4);
  int*   t32i = (int*)alloc((size_t)128 * 4 * 4);
  float* t32w = (float*)alloc((size_t)128 * 4 * 4);
  int*   t64i = (int*)alloc((size_t)128 * 4 * 4);
  float* t64w = (float*)alloc((size_t)128 * 4 * 4);
  if ((size_t)(p - (char*)d_ws) > ws_size) return;  // workspace too small

  // --- tables, weight packs, zero fills ---
  cubic_table<<<1, 128, 0, stream>>>(128, 32, t32i, t32w);
  cubic_table<<<1, 128, 0, stream>>>(128, 64, t64i, t64w);
  pack_w<<<cdiv(9 * 64 * 32, 256), 256, 0, stream>>>(W0, wp0, 64, 3, 32);
  pack_w<<<cdiv(9 * 64 * 64, 256), 256, 0, stream>>>(W2, wp2, 64, 64, 64);
  pack_w<<<cdiv(9 * 128 * 64, 256), 256, 0, stream>>>(W5, wp5, 128, 64, 64);
  pack_w<<<cdiv(9 * 128 * 128, 256), 256, 0, stream>>>(W7, wp7, 128, 128, 128);
  pack_w<<<cdiv(9 * 256 * 128, 256), 256, 0, stream>>>(W10, wp10, 256, 128, 128);
  pack_w<<<cdiv(9 * 256 * 256, 256), 256, 0, stream>>>(W12, wp12, 256, 256, 256);
  pack_w<<<cdiv(9 * 256 * 256, 256), 256, 0, stream>>>(W14, wp14, 256, 256, 256);
  fill_zero<<<cdiv((int)(convZeroBytes / 4), 256), 256, 0, stream>>>(
      (unsigned*)convZeroBase, (int)(convZeroBytes / 4));
  fill_zero<<<cdiv((int)(statsBytes / 4), 256), 256, 0, stream>>>(
      (unsigned*)statsBase, (int)(statsBytes / 4));

  // --- per-image VGG feature pipeline ---
  auto runImage = [&](const float* x, _Float16* normOut) {
    pack_input<<<cdiv(130 * 130 * 32, 256), 256, 0, stream>>>(x, inPad, 128, 128);
    conv3x3_wmma<4><<<dim3(2 * 128, 4), 32, 0, stream>>>(inPad, wp0, B0, f1Pad,
                                                         128, 128, 32, 64, 1);
    conv3x3_wmma<4><<<dim3(2 * 128, 4), 32, 0, stream>>>(f1Pad, wp2, B2, cA,
                                                         128, 128, 64, 64, 1);
    pool2<<<cdiv(64 * 64 * 64, 256), 256, 0, stream>>>(cA, p1, 64, 64, 64);
    conv3x3_wmma<4><<<dim3(64, 8), 32, 0, stream>>>(p1, wp5, B5, cB,
                                                    64, 64, 64, 128, 1);
    conv3x3_wmma<4><<<dim3(64, 8), 32, 0, stream>>>(cB, wp7, B7, cC,
                                                    64, 64, 128, 128, 1);
    pool2<<<cdiv(32 * 32 * 128, 256), 256, 0, stream>>>(cC, p2, 32, 32, 128);
    conv3x3_wmma<2><<<dim3(32, 16), 32, 0, stream>>>(p2, wp10, B10, cD,
                                                     32, 32, 128, 256, 1);
    conv3x3_wmma<2><<<dim3(32, 16), 32, 0, stream>>>(cD, wp12, B12, cE,
                                                     32, 32, 256, 256, 1);
    conv3x3_wmma<2><<<dim3(32, 16), 32, 0, stream>>>(cE, wp14, B14, cF,
                                                     32, 32, 256, 256, 1);
    bicubic_up_feat<<<cdiv(128 * 128 * 32, 256), 256, 0, stream>>>(
        cF, up, t32i, t32w, 32, 256, 128, 128);
    concat_norm<<<cdiv(16384, 128), 128, 0, stream>>>(f1Pad, up, normOut, 128, 128);
  };

  runImage(x_ref, ftemT);  // ftemT used as full-res temp for the reference image
  down2_mean<<<cdiv(4096 * 320, 256), 256, 0, stream>>>(ftemT, frefT);
  runImage(x_tem, ftemT);  // real template features

  // --- correlation: streaming dual softmax + top-5 (3 WMMA passes) ---
  corr_stats<<<dim3(256, 256), 32, 0, stream>>>(frefT, ftemT, coefR, coefT,
                                                M1u, M2u, S1, S2, 0);
  corr_stats<<<dim3(256, 256), 32, 0, stream>>>(frefT, ftemT, coefR, coefT,
                                                M1u, M2u, S1, S2, 1);
  finalize_terms<<<cdiv(16384, 256), 256, 0, stream>>>(M1u, S1, M2u, S2, c1, r2);
  corr_topk<<<256, 128, 0, stream>>>(frefT, ftemT, coefR, coefT, c1, r2, conf);

  // --- head + final upsample ---
  conv_head<<<cdiv(8 * 4096, 256), 256, 0, stream>>>(conf, FW1, FB1, h1,
                                                     5, 8, 64, 64, 3, 1);
  conv_head<<<cdiv(8 * 4096, 256), 256, 0, stream>>>(h1, FW2, FB2, h2,
                                                     8, 8, 64, 64, 3, 1);
  conv_head<<<cdiv(1 * 4096, 256), 256, 0, stream>>>(h2, FW3, FB3, h3,
                                                     8, 1, 64, 64, 1, 0);
  bicubic_up1<<<cdiv(16384, 256), 256, 0, stream>>>(h3, (float*)d_out,
                                                    t64i, t64w, 64, 128, 128);
}
